// GraphTransformerLayer_55121610276984
// MI455X (gfx1250) — matrix operand order repaired and checked by reading, not compiled, not run on hardware
//
#include <hip/hip_runtime.h>
#include <hip/hip_bf16.h>
#include <math.h>

// Graph transformer layer for MI455X (gfx1250), wave32, bf16 WMMA + TDM.
// Pipeline (all on `stream`):
//   0) memsets: acc/denom/mkey zero
//   1) node_proj_wmma : q,k,v,skip = x@W+b   (v_wmma_f32_16x16x32_bf16)
//   2) edge_gemm_wmma : e = edge_attr@We     (TDM tensor_load_to_lds + WMMA), stored bf16
//   3) edge_alpha_max : alpha = <q[dst], k[src]+e>/sqrt(C); atomicMax key per (dst,h)
//   4) edge_accum     : a = exp(alpha-m); atomicAdd denom; atomicAdd acc += a*(v[src]+e)
//   5) node_epilogue  : out = LN(gelu(beta-gated skip) + x)

typedef __attribute__((ext_vector_type(16))) __bf16 v16bf;
typedef __attribute__((ext_vector_type(8)))  float  v8f;
typedef __attribute__((ext_vector_type(4)))  unsigned int v4u;
typedef __attribute__((ext_vector_type(8)))  int    v8i;
typedef __attribute__((ext_vector_type(4)))  int    v4i;

union AB16 { v16bf v; __bf16 e[16]; };
union C8F  { v8f   v; float  e[8];  };

#define D_DIM 128
#define HEADS 4
#define CHAN  32
#define ED_DIM 64

#if __has_include(<hip/amd_detail/amd_gfx1250_TDM.h>)
#define TDM_6ARG 1
#else
#define TDM_6ARG 0
#endif

// ---------------------------------------------------------------- kernel 1
// grid: (ceil(N/16), 8), block: 128 (4 waves). Wave w handles colTile
// ct = blockIdx.y*4+w in [0,32): ct>>3 selects {Wq,Wk,Wv,Wskip}, (ct&7)*16 = col.
__global__ __launch_bounds__(128) void node_proj_wmma(
    const float* __restrict__ x,
    const float* __restrict__ Wq, const float* __restrict__ bq,
    const float* __restrict__ Wk, const float* __restrict__ bk,
    const float* __restrict__ Wv, const float* __restrict__ bv,
    const float* __restrict__ Ws, const float* __restrict__ bs,
    float* __restrict__ qo, float* __restrict__ ko,
    float* __restrict__ vo, float* __restrict__ so, int N)
{
    __shared__ __bf16 xs[16][D_DIM + 8];
    const int tid = threadIdx.x;
    const int nodeBase = blockIdx.x * 16;

    // stage 16 x-rows as bf16 in LDS (coalesced fp32 reads, ds_store_b16)
    for (int r = 0; r < 16; ++r) {
        int row = nodeBase + r; if (row >= N) row = N - 1;   // clamp (reads only)
        xs[r][tid] = (__bf16)x[(size_t)row * D_DIM + tid];
    }
    __syncthreads();

    const int wave = tid >> 5, lane = tid & 31;
    const int half = lane >> 4, mrow = lane & 15;
    const int ct = blockIdx.y * 4 + wave;
    const int colBase = (ct & 7) * 16;

    const float* W; const float* B; float* O;
    switch (ct >> 3) {
        case 0:  W = Wq; B = bq; O = qo; break;
        case 1:  W = Wk; B = bk; O = ko; break;
        case 2:  W = Wv; B = bv; O = vo; break;
        default: W = Ws; B = bs; O = so; break;
    }

    C8F acc;
    #pragma unroll
    for (int i = 0; i < 8; ++i) acc.e[i] = 0.0f;

    #pragma unroll
    for (int k0 = 0; k0 < D_DIM; k0 += 32) {
        if (k0 + 32 < D_DIM)
            __builtin_prefetch(&W[(size_t)(k0 + 32) * D_DIM + colBase + mrow], 0, 0); // global_prefetch_b8
        AB16 a, b;
        #pragma unroll
        for (int i = 0; i < 16; ++i) {
            const int v = i >> 1, s = i & 1;
            const int ka = ((v & 4) ? 16 : 0) + half * 8 + (v & 3) * 2 + s; // A 16x32 bf16 layout
            const int kb = ((v & 4) ? 16 : 0) + half * 8 + (v & 3) * 2 + s; // B 32x16 bf16 layout
            a.e[i] = xs[mrow][k0 + ka];
            b.e[i] = (__bf16)W[(size_t)(k0 + kb) * D_DIM + colBase + mrow];
        }
        acc.v = __builtin_amdgcn_wmma_f32_16x16x32_bf16(false, a.v, false, b.v,
                                                        (short)0, acc.v, false, false);
    }

    const float bias = B[colBase + mrow];
    #pragma unroll
    for (int v = 0; v < 8; ++v) {
        const int row = nodeBase + v + half * 8;                 // C/D: M = v + 8*half
        if (row < N) O[(size_t)row * D_DIM + colBase + mrow] = acc.e[v] + bias;
    }
}

// ---------------------------------------------------------------- kernel 2
// e[E,128] (bf16) = edge_attr[E,64] @ We[64,128].  grid: (ceil(E/16), 2), block 128.
// The 16x64 fp32 A-tile is one contiguous 4KB block of the row-major edge_attr
// matrix -> perfect 1-D Tensor-Data-Mover load with 1-DWORD row padding
// (65-float LDS stride) to avoid bank conflicts on K-column fragment reads.
__global__ __launch_bounds__(128) void edge_gemm_wmma(
    const float* __restrict__ ea, const float* __restrict__ We,
    __bf16* __restrict__ eo, int E)
{
    __shared__ float esf[16][ED_DIM + 1];    // +1: TDM pad_amount = 1 DWORD / 64
    const int tid = threadIdx.x;
    const int eBase = blockIdx.x * 16;

#if __has_builtin(__builtin_amdgcn_tensor_load_to_lds)
    if (tid < 32) {                               // one wave issues the TDM op
        const unsigned lds_off =
            (unsigned)(unsigned long long)(&esf[0][0]);   // LDS addr = addr[31:0]
        const unsigned long long ga =
            (unsigned long long)(const char*)(ea + (size_t)eBase * ED_DIM);
        // ---- D# group 0: count=1 | lds_addr | global_addr | type=2
        v4u g0;
        g0[0] = 1u;                                          // count=1, user mode
        g0[1] = lds_off;                                     // lds_addr
        g0[2] = (unsigned)(ga & 0xFFFFFFFFu);                // global_addr[31:0]
        g0[3] = (unsigned)((ga >> 32) & 0x01FFFFFFu)         // global_addr[56:32]
              | (2u << 30);                                  // type=2 (image)
        // ---- D# group 1: data_size=4B, pad every 64 DW by 1 DW, 1-D tile
        v8i g1;
        g1[0] = (int)((2u << 16)        // data_size = 4 bytes
                    | (1u << 20)        // pad_enable
                    | (5u << 22)        // pad_interval: code 5 = 64 DWORDs
                    | (0u << 25));      // pad_amount:   code 0 = 1 DWORD
        g1[1] = (int)((1024u & 0xFFFFu) << 16);  // tensor_dim0[15:0] = 1024
        g1[2] = (int)(1u << 16);                 // tensor_dim0 hi=0 | tensor_dim1=1
        g1[3] = (int)(1024u << 16);              // tile_dim0 = 1024 elements
        g1[4] = 0;                               // tile_dim1=0 (unused), tile_dim2=0
        g1[5] = 1024;                            // tensor_dim0_stride[31:0]
        g1[6] = 0;
        g1[7] = 0;
        const v4i z4 = {0, 0, 0, 0};
#if TDM_6ARG
        const v8i z8 = {0, 0, 0, 0, 0, 0, 0, 0};
        __builtin_amdgcn_tensor_load_to_lds(g0, g1, z4, z4, z8, 0);  // tensor_load_to_lds
#else
        __builtin_amdgcn_tensor_load_to_lds(g0, g1, z4, z4, 0);      // tensor_load_to_lds
#endif
        __builtin_amdgcn_s_wait_tensorcnt(0);                        // s_wait_tensorcnt 0
    }
#else
    #pragma unroll
    for (int i = 0; i < 8; ++i) {                 // fallback: manual staging
        const int idx = tid + i * 128;
        const int r = idx >> 6, c = idx & 63;
        int row = eBase + r; if (row >= E) row = E - 1;
        esf[r][c] = ea[(size_t)row * ED_DIM + c];
    }
#endif
    __syncthreads();

    const int wave = tid >> 5, lane = tid & 31;
    const int half = lane >> 4, mrow = lane & 15;
    const int colBase = (blockIdx.y * 4 + wave) * 16;

    C8F acc;
    #pragma unroll
    for (int i = 0; i < 8; ++i) acc.e[i] = 0.0f;

    #pragma unroll
    for (int k0 = 0; k0 < ED_DIM; k0 += 32) {
        AB16 a, b;
        #pragma unroll
        for (int i = 0; i < 16; ++i) {
            const int v = i >> 1, s = i & 1;
            const int ka = ((v & 4) ? 16 : 0) + half * 8 + (v & 3) * 2 + s;
            const int kb = ((v & 4) ? 16 : 0) + half * 8 + (v & 3) * 2 + s;
            a.e[i] = (__bf16)esf[mrow][k0 + ka];
            b.e[i] = (__bf16)We[(size_t)(k0 + kb) * D_DIM + colBase + mrow];
        }
        acc.v = __builtin_amdgcn_wmma_f32_16x16x32_bf16(false, a.v, false, b.v,
                                                        (short)0, acc.v, false, false);
    }

    #pragma unroll
    for (int v = 0; v < 8; ++v) {
        const int row = eBase + v + half * 8;
        if (row < E) eo[(size_t)row * D_DIM + colBase + mrow] = (__bf16)acc.e[v];
    }
}

// ---------------------------------------------------------------- kernel 3
// one wave per edge; lane l covers channels [l*4, l*4+4) (head = l>>3).
// alpha = dot(q[dst], k[src]+e)/sqrt(32) per head; atomicMax monotone key.
__global__ __launch_bounds__(256) void edge_alpha_max(
    const int* __restrict__ ei, const float* __restrict__ q,
    const float* __restrict__ kmat, const __bf16* __restrict__ e,
    float* __restrict__ alpha, unsigned int* __restrict__ mkey, int E)
{
    const int wave = threadIdx.x >> 5, lane = threadIdx.x & 31;
    const int edge = blockIdx.x * 8 + wave;
    if (edge >= E) return;
    const int src = ei[edge];
    const int dst = ei[E + edge];
    const int c4 = lane * 4;

    const float4 q4 = *(const float4*)(q    + (size_t)dst * D_DIM + c4);
    const float4 k4 = *(const float4*)(kmat + (size_t)src * D_DIM + c4);
    const __bf16* ep = e + (size_t)edge * D_DIM + c4;

    float p = q4.x * (k4.x + (float)ep[0]) + q4.y * (k4.y + (float)ep[1])
            + q4.z * (k4.z + (float)ep[2]) + q4.w * (k4.w + (float)ep[3]);
    p += __shfl_xor(p, 1, 8);
    p += __shfl_xor(p, 2, 8);
    p += __shfl_xor(p, 4, 8);
    p *= 0.17677669529663687f;            // 1/sqrt(32)

    if ((lane & 7) == 0) {
        const int h = lane >> 3;
        alpha[(size_t)edge * HEADS + h] = p;
        const unsigned int bits = __float_as_uint(p);
        const unsigned int key = (bits & 0x80000000u) ? ~bits : (bits | 0x80000000u);
        atomicMax(&mkey[(size_t)dst * HEADS + h], key);
    }
}

// ---------------------------------------------------------------- kernel 4
__global__ __launch_bounds__(256) void edge_accum(
    const int* __restrict__ ei, const float* __restrict__ vmat,
    const __bf16* __restrict__ e, const float* __restrict__ alpha,
    const unsigned int* __restrict__ mkey,
    float* __restrict__ denom, float* __restrict__ acc, int E)
{
    const int wave = threadIdx.x >> 5, lane = threadIdx.x & 31;
    const int edge = blockIdx.x * 8 + wave;
    if (edge >= E) return;
    const int src = ei[edge];
    const int dst = ei[E + edge];
    const int c4 = lane * 4;
    const int h = lane >> 3;

    const float al = alpha[(size_t)edge * HEADS + h];
    const unsigned int key = mkey[(size_t)dst * HEADS + h];
    float m = 0.0f;
    if (key != 0u) {
        const unsigned int bits = (key & 0x80000000u) ? (key ^ 0x80000000u) : ~key;
        m = __uint_as_float(bits);
        if (!isfinite(m)) m = 0.0f;
    }
    const float a = expf(al - m);
    if ((lane & 7) == 0) atomicAdd(&denom[(size_t)dst * HEADS + h], a);

    const float4 v4 = *(const float4*)(vmat + (size_t)src * D_DIM + c4);
    const __bf16* ep = e + (size_t)edge * D_DIM + c4;
    float* ap = acc + (size_t)dst * D_DIM + c4;
    atomicAdd(ap + 0, a * (v4.x + (float)ep[0]));     // global_atomic_add_f32
    atomicAdd(ap + 1, a * (v4.y + (float)ep[1]));
    atomicAdd(ap + 2, a * (v4.z + (float)ep[2]));
    atomicAdd(ap + 3, a * (v4.w + (float)ep[3]));
}

// ---------------------------------------------------------------- kernel 5
// one wave per node: beta gate (384-dot), exact GELU, residual, LayerNorm(128).
__global__ __launch_bounds__(256) void node_epilogue(
    const float* __restrict__ x, const float* __restrict__ skip,
    const float* __restrict__ acc, const float* __restrict__ denom,
    const float* __restrict__ Wbeta, const float* __restrict__ gamma,
    const float* __restrict__ beta_ln, float* __restrict__ out, int N)
{
    const int wave = threadIdx.x >> 5, lane = threadIdx.x & 31;
    const int node = blockIdx.x * 8 + wave;
    if (node >= N) return;
    const int c4 = lane * 4, h = lane >> 3;

    const float d = denom[(size_t)node * HEADS + h];
    const float inv = 1.0f / (d > 0.0f ? d : 1.0f);

    float o[4], xr[4];
    float s = 0.0f;
    #pragma unroll
    for (int j = 0; j < 4; ++j) {
        o[j]  = acc[(size_t)node * D_DIM + c4 + j] * inv;
        xr[j] = skip[(size_t)node * D_DIM + c4 + j];
        s += o[j] * Wbeta[c4 + j] + xr[j] * Wbeta[D_DIM + c4 + j]
           + (o[j] - xr[j]) * Wbeta[2 * D_DIM + c4 + j];
    }
    #pragma unroll
    for (int off = 1; off < 32; off <<= 1) s += __shfl_xor(s, off, 32); // wave32 reduce
    const float b = 1.0f / (1.0f + expf(-s));

    float g[4];
    float sm = 0.0f;
    #pragma unroll
    for (int j = 0; j < 4; ++j) {
        const float r = b * xr[j] + (1.0f - b) * o[j];
        g[j] = 0.5f * r * (1.0f + erff(r * 0.70710678118654752f))   // exact GELU
             + x[(size_t)node * D_DIM + c4 + j];
        sm += g[j];
    }
    #pragma unroll
    for (int off = 1; off < 32; off <<= 1) sm += __shfl_xor(sm, off, 32);
    const float mu = sm * (1.0f / D_DIM);

    float vs = 0.0f;
    #pragma unroll
    for (int j = 0; j < 4; ++j) { const float t = g[j] - mu; vs += t * t; }
    #pragma unroll
    for (int off = 1; off < 32; off <<= 1) vs += __shfl_xor(vs, off, 32);
    const float rstd = rsqrtf(vs * (1.0f / D_DIM) + 1e-5f);

    #pragma unroll
    for (int j = 0; j < 4; ++j)
        out[(size_t)node * D_DIM + c4 + j] =
            (g[j] - mu) * rstd * gamma[c4 + j] + beta_ln[c4 + j];
}

// ---------------------------------------------------------------- launch
extern "C" void kernel_launch(void* const* d_in, const int* in_sizes, int n_in,
                              void* d_out, int out_size, void* d_ws, size_t ws_size,
                              hipStream_t stream) {
    (void)n_in; (void)out_size; (void)ws_size;
    const float* x       = (const float*)d_in[0];
    const int*   ei      = (const int*)  d_in[1];
    const float* ea      = (const float*)d_in[2];
    const float* Wq      = (const float*)d_in[3];
    const float* bq      = (const float*)d_in[4];
    const float* Wk      = (const float*)d_in[5];
    const float* bk      = (const float*)d_in[6];
    const float* Wv      = (const float*)d_in[7];
    const float* bv      = (const float*)d_in[8];
    const float* We      = (const float*)d_in[9];
    const float* Ws      = (const float*)d_in[10];
    const float* bs      = (const float*)d_in[11];
    const float* Wbeta   = (const float*)d_in[12];
    const float* gamma   = (const float*)d_in[13];
    const float* beta_ln = (const float*)d_in[14];
    float* out = (float*)d_out;

    const int N = in_sizes[0] / D_DIM;       // 50000
    const int E = in_sizes[2] / ED_DIM;      // 800000
    const size_t szNode = (size_t)N * D_DIM; // floats per node matrix

    // workspace carve-up (~348 MB total)
    float* q  = (float*)d_ws;
    float* k  = q  + szNode;
    float* v  = k  + szNode;
    float* sk = v  + szNode;
    float* acc = sk + szNode;
    __bf16* ebf = (__bf16*)(acc + szNode);               // E*128 bf16
    float* alpha = (float*)(ebf + (size_t)E * D_DIM);    // E*4
    unsigned int* mkey = (unsigned int*)(alpha + (size_t)E * HEADS);
    float* denom = (float*)(mkey + (size_t)N * HEADS);

    // deterministic per-call init (graph-capture safe)
    hipMemsetAsync(acc,   0, szNode * sizeof(float),            stream);
    hipMemsetAsync(mkey,  0, (size_t)N * HEADS * sizeof(unsigned int), stream);
    hipMemsetAsync(denom, 0, (size_t)N * HEADS * sizeof(float), stream);

    dim3 gA((N + 15) / 16, 8);
    node_proj_wmma<<<gA, 128, 0, stream>>>(x, Wq, bq, Wk, bk, Wv, bv, Ws, bs,
                                           q, k, v, sk, N);
    dim3 gB((E + 15) / 16, 2);
    edge_gemm_wmma<<<gB, 128, 0, stream>>>(ea, We, ebf, E);

    const int edgeBlocks = (E + 7) / 8;
    edge_alpha_max<<<edgeBlocks, 256, 0, stream>>>(ei, q, k, ebf, alpha, mkey, E);
    edge_accum<<<edgeBlocks, 256, 0, stream>>>(ei, v, ebf, alpha, mkey, denom, acc, E);

    node_epilogue<<<(N + 7) / 8, 256, 0, stream>>>(x, sk, acc, denom,
                                                   Wbeta, gamma, beta_ln, out, N);
}
